// KANLayer_66597762892081
// MI455X (gfx1250) — matrix operand order repaired and checked by reading, not compiled, not run on hardware
//
#include <hip/hip_runtime.h>

typedef __attribute__((ext_vector_type(16))) _Float16 v16h;
typedef __attribute__((ext_vector_type(8)))  _Float16 half8;
typedef __attribute__((ext_vector_type(4)))  _Float16 half4;
typedef __attribute__((ext_vector_type(8)))  float    v8f;

#define BATCH 8192
#define IN_F  1024
#define OUT_F 1024
#define KDIM  2048          // [x | silu(x)] concatenated along K
#define BM    128
#define BN    128
#define BK    32            // one WMMA k-step per LDS tile
#define LDT   40            // padded halfs per LDS row (80B) -> conflict-free b128
#define NK    (KDIM / BK)   // 64
#define NTHR  128           // 4 waves; each wave owns a 64x64 output tile

union FragU { v16h v; half8 h[2]; };

// ---------------- phase 1a: A = [f16(x) | f16(silu(x))], 8 elems/thread ----------------
__global__ __launch_bounds__(256) void kan_prep_act(const float* __restrict__ x,
                                                    _Float16* __restrict__ A) {
  const int t  = blockIdx.x * 256 + threadIdx.x;   // 0 .. BATCH*IN_F/8-1
  const int b  = t >> 7;                           // 128 threads cover one row
  const int i8 = (t & 127) * 8;
  const float4* xp = (const float4*)(x + (size_t)b * IN_F + i8);
  const float4 u0 = xp[0], u1 = xp[1];
  const float vv[8] = {u0.x, u0.y, u0.z, u0.w, u1.x, u1.y, u1.z, u1.w};
  half8 hv, hs;
#pragma unroll
  for (int e = 0; e < 8; ++e) {
    const float v = vv[e];
    hv[e] = (_Float16)v;
    hs[e] = (_Float16)(v / (1.0f + __expf(-v)));   // silu
  }
  _Float16* row = A + (size_t)b * KDIM;
  *(half8*)(row + i8)        = hv;
  *(half8*)(row + IN_F + i8) = hs;
}

// ---------------- phase 1b: W = [f16(W_b) | f16(sum_g W_s)], 4 elems/thread ----------------
__global__ __launch_bounds__(256) void kan_prep_w(const float* __restrict__ bw,
                                                  const float* __restrict__ sw,
                                                  _Float16* __restrict__ W) {
  const int t  = blockIdx.x * 256 + threadIdx.x;   // 0 .. OUT_F*IN_F/4-1
  const int o  = t >> 8;                           // 256 threads cover one row
  const int i4 = (t & 255) * 4;
  const float4 b4 = *(const float4*)(bw + (size_t)o * IN_F + i4);
  const float4* sp = (const float4*)(sw + ((size_t)o * IN_F + i4) * 8);
  half4 hb, hs;
  const float bb[4] = {b4.x, b4.y, b4.z, b4.w};
#pragma unroll
  for (int e = 0; e < 4; ++e) {
    const float4 s0 = sp[e * 2], s1 = sp[e * 2 + 1];
    const float ss = (s0.x + s0.y) + (s0.z + s0.w) + (s1.x + s1.y) + (s1.z + s1.w);
    hb[e] = (_Float16)bb[e];
    hs[e] = (_Float16)ss;
  }
  _Float16* row = W + (size_t)o * KDIM;
  *(half4*)(row + i4)        = hb;
  *(half4*)(row + IN_F + i4) = hs;
}

// ---------------- phase 2: out = A @ W^T via v_wmma_f32_16x16x32_f16 ----------------
__global__ __launch_bounds__(NTHR) void kan_gemm(const _Float16* __restrict__ A,
                                                 const _Float16* __restrict__ W,
                                                 float* __restrict__ out) {
  __shared__ __align__(16) _Float16 smem[2][2][BM * LDT];  // [buf][A/B][row*LDT+k]

  const int tid  = threadIdx.x;
  const int wave = tid >> 5;         // 0..3
  const int lane = tid & 31;
  const int lr   = lane & 15;        // row-in-fragment
  const int lhi  = lane >> 4;        // half-wave select
  const int wm   = wave & 1;         // 2 waves along M (64 rows each)
  const int wn   = wave >> 1;        // 2 waves along N (64 cols each)

  const int row0A = blockIdx.x * BM;
  const int row0B = blockIdx.y * BN;

  // k-advance lives in the (wave-uniform) SGPR base; per-lane voffs stay constant
  uint64_t curA = (uint64_t)(uintptr_t)A;
  uint64_t curW = (uint64_t)(uintptr_t)W;

  // cooperative async-copy geometry: BK=32 halfs/row = 4 chunks of 16B;
  // 128 rows * 4 chunks = 512 chunks; 128 threads -> 4 chunks each (per matrix)
  uint32_t voffA[4], voffB[4];   // constant byte offsets from the advancing bases
  uint32_t lof[4];               // LDS byte offsets within a matrix block
#pragma unroll
  for (int j = 0; j < 4; ++j) {
    const int c   = tid + j * NTHR;
    const int row = c >> 2;
    const int col = (c & 3) * 8;
    voffA[j] = ((uint32_t)(row0A + row) * KDIM + col) * 2u;
    voffB[j] = ((uint32_t)(row0B + row) * KDIM + col) * 2u;
    lof[j]   = (uint32_t)(row * LDT + col) * 2u;
  }

  // issue one BK-slice of async global->LDS copies into buffer `buf`, advance bases
  auto issue = [&](int buf) {
    const uint32_t lbaseA = (uint32_t)(uintptr_t)(void*)&smem[buf][0][0];
    const uint32_t lbaseB = (uint32_t)(uintptr_t)(void*)&smem[buf][1][0];
#pragma unroll
    for (int j = 0; j < 4; ++j) {
      asm volatile("global_load_async_to_lds_b128 %0, %1, %2"
                   :: "v"(lbaseA + lof[j]), "v"(voffA[j]), "s"(curA) : "memory");
      asm volatile("global_load_async_to_lds_b128 %0, %1, %2"
                   :: "v"(lbaseB + lof[j]), "v"(voffB[j]), "s"(curW) : "memory");
    }
    curA += BK * 2;   // 64B per k-slice, scalar adds only
    curW += BK * 2;
  };

  v8f acc[4][4];
#pragma unroll
  for (int i = 0; i < 4; ++i)
#pragma unroll
    for (int j = 0; j < 4; ++j)
#pragma unroll
      for (int e = 0; e < 8; ++e) acc[i][j][e] = 0.0f;

  auto compute = [&](int buf) {
    FragU af[4], bf[4];
    // A fragment (16x32 f16): lane<16 holds K {0..7,16..23}, lane>=16 {8..15,24..31}
#pragma unroll
    for (int i = 0; i < 4; ++i) {
      const _Float16* ap = &smem[buf][0][(wm * 64 + i * 16 + lr) * LDT];
      af[i].h[0] = *(const half8*)(ap + lhi * 8);
      af[i].h[1] = *(const half8*)(ap + 16 + lhi * 8);
    }
    // B fragment (32x16 f16): lane<16 holds K {0..7,16..23}, lane>=16 {8..15,24..31}
#pragma unroll
    for (int j = 0; j < 4; ++j) {
      const _Float16* bp = &smem[buf][1][(wn * 64 + j * 16 + lr) * LDT + lhi * 8];
      bf[j].h[0] = *(const half8*)(bp);
      bf[j].h[1] = *(const half8*)(bp + 16);
    }
#pragma unroll
    for (int i = 0; i < 4; ++i)
#pragma unroll
      for (int j = 0; j < 4; ++j)
        acc[i][j] = __builtin_amdgcn_wmma_f32_16x16x32_f16(
            false, af[i].v, false, bf[j].v, (short)0, acc[i][j], false, false);
  };

  // ---- software pipeline: single barrier per k-step, fully peeled (no conditionals) ----
  issue(0);                                   // k-slice 0 -> buf0
  asm volatile("s_wait_asynccnt 0x0" ::: "memory");
  __syncthreads();

  for (int kt = 0; kt + 2 < NK; kt += 2) {
    issue(1);                                 // k-slice kt+1 -> buf1 (overlaps compute)
    compute(0);
    asm volatile("s_wait_asynccnt 0x0" ::: "memory");
    __syncthreads();

    issue(0);                                 // k-slice kt+2 -> buf0
    compute(1);
    asm volatile("s_wait_asynccnt 0x0" ::: "memory");
    __syncthreads();
  }
  issue(1);                                   // k-slice NK-1 -> buf1
  compute(0);                                 // k-slice NK-2
  asm volatile("s_wait_asynccnt 0x0" ::: "memory");
  __syncthreads();
  compute(1);                                 // k-slice NK-1

  // epilogue: C/D layout: VGPR e, lanes 0-15 -> M=e, lanes 16-31 -> M=e+8; N=lane&15
#pragma unroll
  for (int i = 0; i < 4; ++i) {
#pragma unroll
    for (int j = 0; j < 4; ++j) {
      const int col = row0B + wn * 64 + j * 16 + lr;
      const int row = row0A + wm * 64 + i * 16 + lhi * 8;
      float* op = out + (size_t)row * OUT_F + col;
#pragma unroll
      for (int e = 0; e < 8; ++e)
        op[(size_t)e * OUT_F] = acc[i][j][e];
    }
  }
}

extern "C" void kernel_launch(void* const* d_in, const int* in_sizes, int n_in,
                              void* d_out, int out_size, void* d_ws, size_t ws_size,
                              hipStream_t stream) {
  const float* x  = (const float*)d_in[0];  // [8192,1024]
  const float* bw = (const float*)d_in[1];  // [1024,1024]
  const float* sw = (const float*)d_in[2];  // [1024,1024,8]
  float* out = (float*)d_out;               // [8192,1024]

  _Float16* A = (_Float16*)d_ws;                                     // 32 MB
  _Float16* W = (_Float16*)((char*)d_ws + (size_t)BATCH * KDIM * 2); // +4 MB

  kan_prep_act<<<(BATCH * IN_F / 8) / 256, 256, 0, stream>>>(x, A);
  kan_prep_w<<<(OUT_F * IN_F / 4) / 256, 256, 0, stream>>>(bw, sw, W);

  dim3 grid(BATCH / BM, OUT_F / BN);   // (64, 8)
  kan_gemm<<<grid, NTHR, 0, stream>>>(A, W, out);
}
